// Decoder_1563368096115
// MI455X (gfx1250) — matrix operand order repaired and checked by reading, not compiled, not run on hardware
//
#include <hip/hip_runtime.h>
#include <hip/hip_bf16.h>
#include <math.h>

// ---------------------------------------------------------------------------
// MI455X (gfx1250) persistent GRU-decoder, f16 WMMA (16x16x32) everywhere.
// Round 2: f16 activation mirror (no cvt in hot loop), hoisted A fragments,
// multi-accumulator WMMA chains.
// ---------------------------------------------------------------------------

typedef __attribute__((ext_vector_type(16))) _Float16 v16h;
typedef __attribute__((ext_vector_type(8)))  _Float16 v8h;
typedef __attribute__((ext_vector_type(8)))  float    v8f;
typedef __attribute__((ext_vector_type(4)))  float    v4f;

#define B_TOT   2048
#define T_STEPS 72

// fragment geometry: one B fragment = 32(K) x 16(N) halves = 512 halves
#define FRAG 512

// packed weight tile counts (N-tiles of 16, K-tiles of 32)
#define NT_INIT 8
#define KT_INIT 8
#define NT_G    24
#define KT_WIH  1
#define KT_WHH  4
#define NT_PROJ 16
#define KT_PROJ 4
#define NT_W1   32
#define KT_W1   8
#define NT_W2   16
#define KT_W2   16

// ---------------------------------------------------------------------------
// fragment loaders / stores
// ---------------------------------------------------------------------------

// B fragment: packed contiguously, lane-major: frag[lane*16 + h]
__device__ __forceinline__ v16h load_bfrag(const _Float16* __restrict__ p, int lane) {
    return *(const v16h*)(p + lane * 16);
}

// A fragment from f16 LDS row-major [16][stride]:
// lane L holds row M=L&15; halves 0..7 -> K = kbase + 8*(L>>4) + 0..7
//                          halves 8..15 -> K = kbase + 16 + 8*(L>>4) + 0..7
__device__ __forceinline__ v16h load_afrag16(const _Float16* base, int stride,
                                             int lane, int kbase) {
    const _Float16* p = base + (lane & 15) * stride + kbase + ((lane >> 4) << 3);
    v8h lo = *(const v8h*)p;
    v8h hi = *(const v8h*)(p + 16);
    v16h r;
#pragma unroll
    for (int i = 0; i < 8; ++i) { r[i] = lo[i]; r[i + 8] = hi[i]; }
    return r;
}

// A fragment from f32 LDS row-major (convert on the fly; used once for h0)
__device__ __forceinline__ v16h load_afrag32(const float* base, int stride,
                                             int lane, int kbase) {
    const float* p = base + (lane & 15) * stride + kbase + ((lane >> 4) << 3);
    v4f f0 = *(const v4f*)p;
    v4f f1 = *(const v4f*)(p + 4);
    v4f f2 = *(const v4f*)(p + 16);
    v4f f3 = *(const v4f*)(p + 20);
    v16h r;
#pragma unroll
    for (int i = 0; i < 4; ++i) {
        r[i]      = (_Float16)f0[i];
        r[i + 4]  = (_Float16)f1[i];
        r[i + 8]  = (_Float16)f2[i];
        r[i + 12] = (_Float16)f3[i];
    }
    return r;
}

// C/D 16x16 f32 layout: VGPR r, lane L -> M = r + 8*(L>>4), N = L&15
__device__ __forceinline__ void store_c_f32(float* buf, int stride, int lane,
                                            int col0, const v8f& c, bool add) {
    int n  = col0 + (lane & 15);
    int m0 = (lane >> 4) << 3;
#pragma unroll
    for (int r = 0; r < 8; ++r) {
        float* p = buf + (m0 + r) * stride + n;
        if (add) *p += c[r]; else *p = c[r];
    }
}

__device__ __forceinline__ void store_c_f16(_Float16* buf, int stride, int lane,
                                            int col0, const v8f& c) {
    int n  = col0 + (lane & 15);
    int m0 = (lane >> 4) << 3;
#pragma unroll
    for (int r = 0; r < 8; ++r)
        buf[(m0 + r) * stride + n] = (_Float16)c[r];
}

__device__ __forceinline__ v8f wmma_acc(const v8f& c, const v16h& a, const v16h& b) {
    return __builtin_amdgcn_wmma_f32_16x16x32_f16(
        /*neg_a=*/false, a, /*neg_b=*/false, b,
        /*c_mod=*/(short)0, c, /*reuse_a=*/false, /*reuse_b=*/false);
}

__device__ __forceinline__ v8f bcast8(float v) {
    v8f c;
#pragma unroll
    for (int r = 0; r < 8; ++r) c[r] = v;
    return c;
}

__device__ __forceinline__ float sigm(float x) { return 1.0f / (1.0f + expf(-x)); }
__device__ __forceinline__ float gelu_exact(float x) {
    return 0.5f * x * (1.0f + erff(x * 0.70710678118654752f));
}

// ---------------------------------------------------------------------------
// weight packer: row-major f32 W[N][K] -> B fragments (nt-major, then kt):
// dst[((nt*KT+kt)*32 + lane)*16 + h] = W[nt*16+(lane&15)][kt*32 + (h&7) + 16*(h>>3) + 8*(lane>>4)]
// ---------------------------------------------------------------------------
__global__ void pack_kernel(const float* __restrict__ W, _Float16* __restrict__ dst,
                            int N, int K, int NT, int KT) {
    int idx   = blockIdx.x * 256 + threadIdx.x;
    int total = NT * KT * FRAG;
    if (idx >= total) return;
    int h    = idx & 15;
    int lane = (idx >> 4) & 31;
    int frag = idx >> 9;
    int kt   = frag % KT;
    int nt   = frag / KT;
    int n    = nt * 16 + (lane & 15);
    int k    = kt * 32 + (h & 7) + ((h >> 3) << 4) + ((lane >> 4) << 3);
    float v  = (n < N && k < K) ? W[(size_t)n * K + k] : 0.0f;
    dst[idx] = (_Float16)v;
}

// ---------------------------------------------------------------------------
// main persistent decoder kernel: 128 blocks x 256 threads (8 waves).
// Each block owns batch rows [bm, bm+16) for all 72 steps.
//
// LDS map (46336 B):
//   h16  [16][128] f16      @ 0      (4096)
//   in16 [16][32]  f16      @ 4096   (1024)
//   x16  [16][256] f16      @ 5120   (8192)
//   regA                    @ 13312  (16384): y16 [16][512] f16
//                                     alias  gxn [16][128] f32 + ghn [16][128] f32
//   regB                    @ 29696  (16384): xf [16][256] f32  alias grz [16][256] f32
//   posb [16][2]  f32       @ 46080  (128)
//   vprv [16][2]  f32       @ 46208  (128)
// ---------------------------------------------------------------------------
__global__ __launch_bounds__(256) void decoder_kernel(
    const float* __restrict__ cond, const float* __restrict__ z,
    const float* __restrict__ last_pos,
    const float* __restrict__ init_b,
    const float* __restrict__ bih,  const float* __restrict__ bhh,
    const float* __restrict__ proj_b,
    const float* __restrict__ b1,   const float* __restrict__ b2,
    const float* __restrict__ lnw,  const float* __restrict__ lnb,
    const float* __restrict__ muw,  const float* __restrict__ mub,
    const float* __restrict__ lvw,  const float* __restrict__ lvb,
    const float* __restrict__ v0,
    const _Float16* __restrict__ p_init, const _Float16* __restrict__ p_wih,
    const _Float16* __restrict__ p_whh,  const _Float16* __restrict__ p_proj,
    const _Float16* __restrict__ p_w1,   const _Float16* __restrict__ p_w2,
    float* __restrict__ out) {

    __shared__ __align__(16) unsigned char smem[46336];
    _Float16* h16  = (_Float16*)(smem);
    _Float16* in16 = (_Float16*)(smem + 4096);
    _Float16* x16  = (_Float16*)(smem + 5120);
    _Float16* y16  = (_Float16*)(smem + 13312);
    float*    gxn  = (float*)(smem + 13312);
    float*    ghn  = (float*)(smem + 21504);
    float*    xf   = (float*)(smem + 29696);
    float*    grz  = (float*)(smem + 29696);
    float*    posb = (float*)(smem + 46080);
    float*    vprv = (float*)(smem + 46208);

    const int tid  = threadIdx.x;
    const int wave = tid >> 5;
    const int lane = tid & 31;
    const int bm   = blockIdx.x * 16;

    float* out_mu = out;
    float* out_lv = out + (size_t)B_TOT * T_STEPS * 2;
    float* out_dl = out + (size_t)2 * B_TOT * T_STEPS * 2;

    // ---- init: z into in16 cols 2..17, zero pad 18..31; pos/vprev; cond -> xf
    for (int e = tid; e < 16 * 32; e += 256) {
        int m = e >> 5, c = e & 31;
        float v = 0.0f;
        if (c >= 2 && c < 18) v = z[(size_t)(bm + m) * 16 + (c - 2)];
        in16[e] = (_Float16)v;
    }
    if (tid < 32) {
        int m = tid >> 1, o = tid & 1;
        posb[m * 2 + o] = last_pos[(size_t)(bm + m) * 2 + o];
        vprv[m * 2 + o] = v0[o];
    }
    for (int e = tid; e < 16 * 256; e += 256) {
        int m = e >> 8, c = e & 255;
        xf[e] = cond[(size_t)(bm + m) * 256 + c];
    }
    __syncthreads();

    // ---- h0 = tanh(cond @ init_w.T + init_b): N=128 (1 ntile/wave), K=256
    {
        int nt = wave;
        int n  = nt * 16 + (lane & 15);
        v8f acc = bcast8(init_b[n]);
        for (int kt = 0; kt < KT_INIT; ++kt) {
            v16h a = load_afrag32(xf, 256, lane, kt * 32);
            v16h b = load_bfrag(p_init + (size_t)(nt * KT_INIT + kt) * FRAG, lane);
            acc = wmma_acc(acc, a, b);
        }
#pragma unroll
        for (int r = 0; r < 8; ++r) acc[r] = tanhf(acc[r]);
        store_c_f16(h16, 128, lane, nt * 16, acc);
    }
    __syncthreads();

    // ======================= time loop =======================
    for (int t = 0; t < T_STEPS; ++t) {
        // v_prev -> in16 cols 0,1
        if (tid < 32) {
            int m = tid >> 1, o = tid & 1;
            in16[m * 32 + o] = (_Float16)vprv[m * 2 + o];
        }
        __syncthreads();

        // ---- GRU gates: N=384 -> 24 ntiles, 3 per wave.
        //      nt<16  (r,z): single accumulator gx+gh  -> grz
        //      nt>=16 (n)  : gx -> gxn, gh -> ghn separately
        {
            const int nt0 = wave * 3;
            v8f acc[3];
#pragma unroll
            for (int j = 0; j < 3; ++j) {
                int nt = nt0 + j, n = nt * 16 + (lane & 15);
                acc[j] = bcast8(bih[n] + ((nt < 16) ? bhh[n] : 0.0f));
            }
            {   // x-path: K=18 (padded to 32), A = in16 (shared across ntiles)
                v16h a = load_afrag16(in16, 32, lane, 0);
#pragma unroll
                for (int j = 0; j < 3; ++j) {
                    v16h b = load_bfrag(p_wih + (size_t)(nt0 + j) * FRAG, lane);
                    acc[j] = wmma_acc(acc[j], a, b);
                }
            }
#pragma unroll
            for (int j = 0; j < 3; ++j) {
                int nt = nt0 + j;
                if (nt >= 16) {   // split mode: flush gx, restart with bhh
                    int n = nt * 16 + (lane & 15);
                    store_c_f32(gxn, 128, lane, (nt - 16) * 16, acc[j], false);
                    acc[j] = bcast8(bhh[n]);
                }
            }
            for (int kt = 0; kt < KT_WHH; ++kt) {   // h-path: K=128, A = h16
                v16h a = load_afrag16(h16, 128, lane, kt * 32);
#pragma unroll
                for (int j = 0; j < 3; ++j) {
                    v16h b = load_bfrag(
                        p_whh + (size_t)((nt0 + j) * KT_WHH + kt) * FRAG, lane);
                    acc[j] = wmma_acc(acc[j], a, b);
                }
            }
#pragma unroll
            for (int j = 0; j < 3; ++j) {
                int nt = nt0 + j;
                if (nt < 16) store_c_f32(grz, 256, lane, nt * 16, acc[j], false);
                else         store_c_f32(ghn, 128, lane, (nt - 16) * 16, acc[j], false);
            }
        }
        __syncthreads();

        // ---- gate nonlinearity + h update (elementwise, 16x128)
        for (int e = tid; e < 16 * 128; e += 256) {
            int m = e >> 7, j = e & 127;
            float r  = sigm(grz[m * 256 + j]);
            float zg = sigm(grz[m * 256 + 128 + j]);
            float nn = tanhf(gxn[m * 128 + j] + r * ghn[m * 128 + j]);
            float hp = (float)h16[m * 128 + j];
            h16[m * 128 + j] = (_Float16)((1.0f - zg) * nn + zg * hp);
        }
        __syncthreads();

        // ---- head projection: x = h @ proj_w.T + proj_b  (N=256, K=128)
        {
            const int nt0 = wave * 2;
            v8f acc[2];
#pragma unroll
            for (int j = 0; j < 2; ++j)
                acc[j] = bcast8(proj_b[(nt0 + j) * 16 + (lane & 15)]);
            for (int kt = 0; kt < KT_PROJ; ++kt) {
                v16h a = load_afrag16(h16, 128, lane, kt * 32);
#pragma unroll
                for (int j = 0; j < 2; ++j) {
                    v16h b = load_bfrag(
                        p_proj + (size_t)((nt0 + j) * KT_PROJ + kt) * FRAG, lane);
                    acc[j] = wmma_acc(acc[j], a, b);
                }
            }
#pragma unroll
            for (int j = 0; j < 2; ++j) {
                store_c_f32(xf, 256, lane, (nt0 + j) * 16, acc[j], false);
                store_c_f16(x16, 256, lane, (nt0 + j) * 16, acc[j]);
            }
        }
        __syncthreads();

        // ---- 3 residual blocks
        for (int i = 0; i < 3; ++i) {
            const _Float16* W1 = p_w1 + (size_t)i * NT_W1 * KT_W1 * FRAG;
            const _Float16* W2 = p_w2 + (size_t)i * NT_W2 * KT_W2 * FRAG;

            // y = gelu(x @ w1.T + b1): N=512 (4 ntiles/wave), K=256, A = x16
            {
                const int nt0 = wave * 4;
                v8f acc[4];
#pragma unroll
                for (int j = 0; j < 4; ++j)
                    acc[j] = bcast8(b1[i * 512 + (nt0 + j) * 16 + (lane & 15)]);
                for (int kt = 0; kt < KT_W1; ++kt) {
                    v16h a = load_afrag16(x16, 256, lane, kt * 32);
#pragma unroll
                    for (int j = 0; j < 4; ++j) {
                        v16h b = load_bfrag(
                            W1 + (size_t)((nt0 + j) * KT_W1 + kt) * FRAG, lane);
                        acc[j] = wmma_acc(acc[j], a, b);
                    }
                }
#pragma unroll
                for (int j = 0; j < 4; ++j) {
#pragma unroll
                    for (int r = 0; r < 8; ++r) acc[j][r] = gelu_exact(acc[j][r]);
                    store_c_f16(y16, 512, lane, (nt0 + j) * 16, acc[j]);
                }
            }
            __syncthreads();

            // x += y @ w2.T + b2: N=256 (2 ntiles/wave), K=512, A = y16
            {
                const int nt0 = wave * 2;
                v8f acc[2];
#pragma unroll
                for (int j = 0; j < 2; ++j)
                    acc[j] = bcast8(b2[i * 256 + (nt0 + j) * 16 + (lane & 15)]);
                for (int kt = 0; kt < KT_W2; ++kt) {
                    v16h a = load_afrag16(y16, 512, lane, kt * 32);
#pragma unroll
                    for (int j = 0; j < 2; ++j) {
                        v16h b = load_bfrag(
                            W2 + (size_t)((nt0 + j) * KT_W2 + kt) * FRAG, lane);
                        acc[j] = wmma_acc(acc[j], a, b);
                    }
                }
#pragma unroll
                for (int j = 0; j < 2; ++j)
                    store_c_f32(xf, 256, lane, (nt0 + j) * 16, acc[j], true);
            }
            __syncthreads();

            // layernorm per row: 16 lanes per row, shuffle reduce (width 16);
            // writes both xf (f32 master) and x16 (f16 mirror for next GEMM)
            {
                int m = tid >> 4, li = tid & 15;
                float s = 0.0f, ss = 0.0f;
#pragma unroll
                for (int k = 0; k < 16; ++k) {
                    float v = xf[m * 256 + li + 16 * k];
                    s += v; ss += v * v;
                }
                for (int o = 1; o < 16; o <<= 1) {
                    s  += __shfl_xor(s,  o, 16);
                    ss += __shfl_xor(ss, o, 16);
                }
                float mean = s * (1.0f / 256.0f);
                float var  = ss * (1.0f / 256.0f) - mean * mean;
                float rstd = rsqrtf(var + 1e-5f);
#pragma unroll
                for (int k = 0; k < 16; ++k) {
                    int c = li + 16 * k;
                    float v = xf[m * 256 + c];
                    float o2 = (v - mean) * rstd * lnw[i * 256 + c] + lnb[i * 256 + c];
                    xf[m * 256 + c]  = o2;
                    x16[m * 256 + c] = (_Float16)o2;
                }
            }
            __syncthreads();
        }

        // ---- output heads (OUT=2): wave0 -> mu/pos/delta, wave1 -> logvar
        if (wave == 0) {
            int m = lane & 15, o = lane >> 4;
            const v4f* xr = (const v4f*)(xf + m * 256);
            const v4f* wr = (const v4f*)(muw + (size_t)o * 256);
            float d = mub[o];
#pragma unroll 4
            for (int j = 0; j < 64; ++j) {
                v4f a = xr[j], w = wr[j];
                d += a[0] * w[0] + a[1] * w[1] + a[2] * w[2] + a[3] * w[3];
            }
            float p = posb[m * 2 + o] + d;
            posb[m * 2 + o] = p;
            vprv[m * 2 + o] = d;
            size_t oi = ((size_t)(bm + m) * T_STEPS + t) * 2 + o;
            out_mu[oi] = p;
            out_dl[oi] = d;
        } else if (wave == 1) {
            int m = lane & 15, o = lane >> 4;
            const v4f* xr = (const v4f*)(xf + m * 256);
            const v4f* wr = (const v4f*)(lvw + (size_t)o * 256);
            float raw = lvb[o];
#pragma unroll 4
            for (int j = 0; j < 64; ++j) {
                v4f a = xr[j], w = wr[j];
                raw += a[0] * w[0] + a[1] * w[1] + a[2] * w[2] + a[3] * w[3];
            }
            raw = fminf(fmaxf(raw, -10.0f), 10.0f);
            float var = fminf(log1pf(expf(raw)) + 0.01f, 10.0f);
            size_t oi = ((size_t)(bm + m) * T_STEPS + t) * 2 + o;
            out_lv[oi] = logf(var);
        }
        __syncthreads();
    }
}

// ---------------------------------------------------------------------------
// host launcher
// ---------------------------------------------------------------------------
extern "C" void kernel_launch(void* const* d_in, const int* in_sizes, int n_in,
                              void* d_out, int out_size, void* d_ws, size_t ws_size,
                              hipStream_t stream) {
    const float* cond     = (const float*)d_in[0];
    const float* z        = (const float*)d_in[1];
    const float* last_pos = (const float*)d_in[2];
    const float* init_w   = (const float*)d_in[3];
    const float* init_b   = (const float*)d_in[4];
    const float* gru_wih  = (const float*)d_in[5];
    const float* gru_whh  = (const float*)d_in[6];
    const float* gru_bih  = (const float*)d_in[7];
    const float* gru_bhh  = (const float*)d_in[8];
    const float* proj_w   = (const float*)d_in[9];
    const float* proj_b   = (const float*)d_in[10];
    const float* blk_w1   = (const float*)d_in[11];
    const float* blk_b1   = (const float*)d_in[12];
    const float* blk_w2   = (const float*)d_in[13];
    const float* blk_b2   = (const float*)d_in[14];
    const float* ln_w     = (const float*)d_in[15];
    const float* ln_b     = (const float*)d_in[16];
    const float* mu_w     = (const float*)d_in[17];
    const float* mu_b     = (const float*)d_in[18];
    const float* lv_w     = (const float*)d_in[19];
    const float* lv_b     = (const float*)d_in[20];
    const float* v0       = (const float*)d_in[21];

    _Float16* ws = (_Float16*)d_ws;
    size_t o_init = 0;
    size_t o_wih  = o_init + (size_t)NT_INIT * KT_INIT * FRAG;
    size_t o_whh  = o_wih  + (size_t)NT_G    * KT_WIH  * FRAG;
    size_t o_proj = o_whh  + (size_t)NT_G    * KT_WHH  * FRAG;
    size_t o_w1   = o_proj + (size_t)NT_PROJ * KT_PROJ * FRAG;
    size_t o_w2   = o_w1   + (size_t)3 * NT_W1 * KT_W1 * FRAG;

    auto launch_pack = [&](const float* W, _Float16* dst, int N, int K, int NT, int KT) {
        int total = NT * KT * FRAG;
        pack_kernel<<<(total + 255) / 256, 256, 0, stream>>>(W, dst, N, K, NT, KT);
    };

    launch_pack(init_w,  ws + o_init, 128, 256, NT_INIT, KT_INIT);
    launch_pack(gru_wih, ws + o_wih,  384, 18,  NT_G,    KT_WIH);
    launch_pack(gru_whh, ws + o_whh,  384, 128, NT_G,    KT_WHH);
    launch_pack(proj_w,  ws + o_proj, 256, 128, NT_PROJ, KT_PROJ);
    for (int i = 0; i < 3; ++i) {
        launch_pack(blk_w1 + (size_t)i * 512 * 256,
                    ws + o_w1 + (size_t)i * NT_W1 * KT_W1 * FRAG, 512, 256, NT_W1, KT_W1);
        launch_pack(blk_w2 + (size_t)i * 256 * 512,
                    ws + o_w2 + (size_t)i * NT_W2 * KT_W2 * FRAG, 256, 512, NT_W2, KT_W2);
    }

    decoder_kernel<<<B_TOT / 16, 256, 0, stream>>>(
        cond, z, last_pos, init_b, gru_bih, gru_bhh, proj_b,
        blk_b1, blk_b2, ln_w, ln_b, mu_w, mu_b, lv_w, lv_b, v0,
        ws + o_init, ws + o_wih, ws + o_whh, ws + o_proj, ws + o_w1, ws + o_w2,
        (float*)d_out);
}
